// Hgru1dModule_62603443306629
// MI455X (gfx1250) — matrix operand order repaired and checked by reading, not compiled, not run on hardware
//
#include <hip/hip_runtime.h>
#include <hip/hip_bf16.h>

// ---------------------------------------------------------------------------
// Hgru1d for MI455X (gfx1250): bf16 WMMA GEMMs with async-to-LDS double
// buffering + chunk-parallel bidirectional linear scan.
//   B=16, N=4096, D=512  (M = B*N = 65536 rows)
// Pipeline:
//   0. cvt: x -> xb (bf16), W_in/W_gate/W_out -> bf16 (one cheap pass each)
//   1. gemm<0>: uf = xb@W_in (+b) -> fused epilogue writes lam, v  (WMMA bf16)
//   2. scan_pass1/2/3: bidirectional h_t = lam*h + v (chunked, 3-pass)
//   3. gemm<1>: g = sigmoid(xb@W_gate + b) (written into dead lam buffer)
//   4. norm_gate: h * rsqrt(mean(h^2)+eps) * g -> hn (bf16)
//   5. gemm<2>: out = hn@W_out + b_out
// GEMM A-tiles are staged with GLOBAL_LOAD_ASYNC_TO_LDS_B128 (ASYNCcnt) into
// double-buffered LDS; compute on tile k overlaps staging of tile k+1.
// Workspace use: ~522 MiB of d_ws.
// ---------------------------------------------------------------------------

typedef __bf16 bf16_t;
typedef __attribute__((ext_vector_type(16))) __bf16        v16bf;
typedef __attribute__((ext_vector_type(4)))  __bf16        v4bf;
typedef __attribute__((ext_vector_type(8)))  float         v8f;
typedef __attribute__((ext_vector_type(8)))  unsigned int  v8u;
typedef __attribute__((ext_vector_type(4)))  unsigned int  uv4;

#define D_DIM   512
#define N_DIM   4096
#define B_DIM   16
#define M_DIM   (B_DIM * N_DIM)   // 65536
#define NCH     32
#define CHLEN   (N_DIM / NCH)     // 128
#define LDA     40                // padded LDS row stride (bf16), 80 B = 16 B-aligned

__device__ __forceinline__ unsigned short f2bf(float f) {
    unsigned int u = __builtin_bit_cast(unsigned int, f);
    unsigned int r = u + 0x7FFFu + ((u >> 16) & 1u);   // round-to-nearest-even
    return (unsigned short)(r >> 16);
}
__device__ __forceinline__ unsigned int pack2bf(float a, float b) {
    return (unsigned int)f2bf(a) | ((unsigned int)f2bf(b) << 16);
}
__device__ __forceinline__ float sigmoid_f(float x) {
    return 1.0f / (1.0f + __expf(-x));
}
__device__ __forceinline__ v16bf load_frag(const bf16_t* p_lo, const bf16_t* p_hi) {
    uv4 lo = *(const uv4*)p_lo;
    uv4 hi = *(const uv4*)p_hi;
    v8u t;
    t[0] = lo[0]; t[1] = lo[1]; t[2] = lo[2]; t[3] = lo[3];
    t[4] = hi[0]; t[5] = hi[1]; t[6] = hi[2]; t[7] = hi[3];
    return __builtin_bit_cast(v16bf, t);
}
// 16-byte async copy global(bf16) -> LDS; tracked by ASYNCcnt.
// Low 32 bits of a flat pointer into LDS are the wave-relative LDS address.
__device__ __forceinline__ void async_copy_b128(const bf16_t* gsrc, bf16_t* ldst) {
    unsigned int       laddr = (unsigned int)(unsigned long long)(const void*)ldst;
    unsigned long long gaddr = (unsigned long long)gsrc;
    asm volatile("global_load_async_to_lds_b128 %0, %1, off"
                 :: "v"(laddr), "v"(gaddr) : "memory");
}
__device__ __forceinline__ void wait_async0() {
    asm volatile("s_wait_asynccnt 0x0" ::: "memory");
}

// ---------------------------------------------------------------------------
// fp32 -> bf16 bulk convert (float4 -> 2x packed uint)
// ---------------------------------------------------------------------------
__global__ __launch_bounds__(256)
void cvt_f32_bf16_kernel(const float* __restrict__ s, unsigned int* __restrict__ d, int n4)
{
    int i = blockIdx.x * blockDim.x + threadIdx.x;
    if (i < n4) {
        float4 f = ((const float4*)s)[i];
        d[i * 2]     = pack2bf(f.x, f.y);
        d[i * 2 + 1] = pack2bf(f.z, f.w);
    }
}

// ---------------------------------------------------------------------------
// WMMA GEMM:  C(M x 512) = A(M x 512, bf16) * W(512 x ldb, bf16)[:, slice] + bias
// MODE 0: W=W_in (ldb=1024): u-cols [n], f-cols [512+n];
//         epilogue lam = lb+(1-lb)*sig(f) ; v = (1-lam)*silu(u)
// MODE 1: W=W_gate (ldb=512): epilogue g = sigmoid(.)
// MODE 2: A=hn, W=W_out (ldb=512): plain + bias -> out
// Block: 256 threads = 8 waves (4 over M x 2 over N); tile 128x64, K-step 32.
// Double-buffered LDS; A staged via async-to-LDS, B transposed synchronously.
// ---------------------------------------------------------------------------
template <int MODE>
__global__ __launch_bounds__(256)
void gemm_wmma_kernel(const bf16_t* __restrict__ A, const bf16_t* __restrict__ W,
                      const float* __restrict__ bias, const float* __restrict__ lbp,
                      float* __restrict__ out0, float* __restrict__ out1,
                      int K, int ldb)
{
    __shared__ bf16_t As[2][128 * LDA];
    __shared__ bf16_t Bs[2][2][64 * LDA];

    const int tid  = threadIdx.x;
    const int lane = tid & 31;
    const int w    = tid >> 5;
    const int wm   = w & 3;    // wave row 0..3  (32 rows each)
    const int wn   = w >> 2;   // wave col 0..1  (32 cols each)
    const int n0   = blockIdx.x * 64;
    const int m0   = blockIdx.y * 128;

    v8f accU[2][2] = {};
    v8f accF[2][2] = {};

    const int arow = tid >> 2;         // 0..63 (A copy: 4 chunks/row)
    const int acol = (tid & 3) * 8;    // bf16 col: 0,8,16,24 -> 16B chunks
    const int bkk  = tid >> 4;         // 0..15 (B copy: k row)
    const int bnn  = (tid & 15) * 4;   // 0..60 (B copy: 4 cols = 8B load)

    auto stage = [&](int buf, int k0) {
        // ---- A tile (128 x 32 bf16 = 8 KB): async 16B copies, 2 per thread ----
#pragma unroll
        for (int it = 0; it < 2; ++it) {
            int row = arow + it * 64;
            async_copy_b128(A + (size_t)(m0 + row) * K + k0 + acol,
                            &As[buf][row * LDA + acol]);
        }
        // ---- B tile(s) (32 x 64) transposed into Bs[n][k] (L2-resident) ----
        const int nMats = (MODE == 0) ? 2 : 1;
#pragma unroll
        for (int mat = 0; mat < nMats; ++mat) {
            int colBase = n0 + (mat ? 512 : 0);
#pragma unroll
            for (int it = 0; it < 2; ++it) {
                int kk = bkk + it * 16;   // 0..31
                v4bf q = *(const v4bf*)(W + (size_t)(k0 + kk) * ldb + colBase + bnn);
#pragma unroll
                for (int e = 0; e < 4; ++e)
                    Bs[buf][mat][(bnn + e) * LDA + kk] = q[e];
            }
        }
    };

    stage(0, 0);
    wait_async0();
    __syncthreads();

    const int rsel = lane & 15;
    const int klo  = (lane >> 4) * 8;    // A: K halves 0..7 / 8..15 (+16)
    const int kb   = (lane >> 4) * 8;    // B: K halves 0..7 / 8..15 (+16)

    int buf = 0;
    for (int k0 = 0; k0 < K; k0 += 32) {
        const int nxt = buf ^ 1;
        if (k0 + 32 < K) stage(nxt, k0 + 32);   // overlap with WMMA below

        v16bf aF[2], bU[2], bF[2];
#pragma unroll
        for (int i = 0; i < 2; ++i) {
            int row = wm * 32 + i * 16 + rsel;
            aF[i] = load_frag(&As[buf][row * LDA + klo],
                              &As[buf][row * LDA + 16 + klo]);
        }
#pragma unroll
        for (int j = 0; j < 2; ++j) {
            int col = wn * 32 + j * 16 + rsel;
            bU[j] = load_frag(&Bs[buf][0][col * LDA + kb],
                              &Bs[buf][0][col * LDA + 16 + kb]);
            if (MODE == 0)
                bF[j] = load_frag(&Bs[buf][1][col * LDA + kb],
                                  &Bs[buf][1][col * LDA + 16 + kb]);
        }
#pragma unroll
        for (int i = 0; i < 2; ++i)
#pragma unroll
            for (int j = 0; j < 2; ++j) {
                accU[i][j] = __builtin_amdgcn_wmma_f32_16x16x32_bf16(
                    false, aF[i], false, bU[j], (short)0, accU[i][j], false, false);
                if (MODE == 0)
                    accF[i][j] = __builtin_amdgcn_wmma_f32_16x16x32_bf16(
                        false, aF[i], false, bF[j], (short)0, accF[i][j], false, false);
            }

        if (k0 + 32 < K) wait_async0();
        __syncthreads();
        buf = nxt;
    }

    // ---- epilogue (C/D layout: col = lane&15, VGPR r -> M = r + 8*(lane>>4)) ----
    const float lb = (MODE == 0) ? lbp[0] : 0.0f;
#pragma unroll
    for (int i = 0; i < 2; ++i)
#pragma unroll
        for (int j = 0; j < 2; ++j) {
            int gcol    = n0 + wn * 32 + j * 16 + (lane & 15);
            int rowBase = m0 + wm * 32 + i * 16 + ((lane >> 4) * 8);
            float bu  = bias[gcol];
            float bf_ = (MODE == 0) ? bias[512 + gcol] : 0.0f;
#pragma unroll
            for (int r = 0; r < 8; ++r) {
                size_t idx = (size_t)(rowBase + r) * D_DIM + gcol;
                float cu = accU[i][j][r] + bu;
                if (MODE == 0) {
                    float cf  = accF[i][j][r] + bf_;
                    float su  = cu * sigmoid_f(cu);                 // silu(u)
                    float lam = lb + (1.0f - lb) * sigmoid_f(cf);   // bounded gate
                    out0[idx] = lam;
                    out1[idx] = (1.0f - lam) * su;                  // v
                } else if (MODE == 1) {
                    out0[idx] = sigmoid_f(cu);                      // g
                } else {
                    out0[idx] = cu;                                 // final out
                }
            }
        }
}

// ---------------------------------------------------------------------------
// Chunk-parallel bidirectional linear recurrence (3-pass).
// ---------------------------------------------------------------------------
__global__ __launch_bounds__(256)
void scan_pass1(const float* __restrict__ lam, const float* __restrict__ v,
                float* __restrict__ P, float* __restrict__ S)
{
    int idx = blockIdx.x * blockDim.x + threadIdx.x;  // B*NCH*D = 262144
    int dir = blockIdx.y;
    int d = idx & 511;
    int c = (idx >> 9) & 31;
    int b = idx >> 14;
    size_t base = ((size_t)b * N_DIM + (size_t)c * CHLEN) * D_DIM + d;
    float p = 1.0f, h = 0.0f;
    if (dir == 0) {
        for (int t = 0; t < CHLEN; ++t) {
            size_t o = base + (size_t)t * D_DIM;
            float l = lam[o];
            h = fmaf(l, h, v[o]);
            p *= l;
        }
    } else {
        for (int t = CHLEN - 1; t >= 0; --t) {
            size_t o = base + (size_t)t * D_DIM;
            float l = lam[o];
            h = fmaf(l, h, v[o]);
            p *= l;
        }
    }
    size_t so = (((size_t)b * 2 + dir) * NCH + c) * D_DIM + d;
    P[so] = p;
    S[so] = h;
}

__global__ __launch_bounds__(256)
void scan_pass2(const float* __restrict__ P, const float* __restrict__ S,
                float* __restrict__ Carr)
{
    int idx = blockIdx.x * blockDim.x + threadIdx.x;  // B*D*2 = 16384
    int d   = idx & 511;
    int b   = (idx >> 9) & 15;
    int dir = idx >> 13;
    float carry = 0.0f;
    if (dir == 0) {
        for (int c = 0; c < NCH; ++c) {
            size_t o = (((size_t)b * 2 + 0) * NCH + c) * D_DIM + d;
            Carr[o] = carry;
            carry = fmaf(P[o], carry, S[o]);
        }
    } else {
        for (int c = NCH - 1; c >= 0; --c) {
            size_t o = (((size_t)b * 2 + 1) * NCH + c) * D_DIM + d;
            Carr[o] = carry;
            carry = fmaf(P[o], carry, S[o]);
        }
    }
}

__global__ __launch_bounds__(256)
void scan_pass3(const float* __restrict__ lam, const float* __restrict__ v,
                const float* __restrict__ Carr, float* __restrict__ h)
{
    int idx = blockIdx.x * blockDim.x + threadIdx.x;  // B*NCH*D = 262144
    int d = idx & 511;
    int c = (idx >> 9) & 31;
    int b = idx >> 14;
    size_t base = ((size_t)b * N_DIM + (size_t)c * CHLEN) * D_DIM + d;
    size_t cf = (((size_t)b * 2 + 0) * NCH + c) * D_DIM + d;
    size_t cb = (((size_t)b * 2 + 1) * NCH + c) * D_DIM + d;
    float hf = Carr[cf];
    for (int t = 0; t < CHLEN; ++t) {
        size_t o = base + (size_t)t * D_DIM;
        hf = fmaf(lam[o], hf, v[o]);
        h[o] = hf;
    }
    float hb = Carr[cb];
    for (int t = CHLEN - 1; t >= 0; --t) {
        size_t o = base + (size_t)t * D_DIM;
        hb = fmaf(lam[o], hb, v[o]);
        h[o] += hb;
    }
}

// ---------------------------------------------------------------------------
// SimpleRMSNorm over D + output gate; writes hn in bf16 for the final GEMM.
// ---------------------------------------------------------------------------
__global__ __launch_bounds__(256)
void norm_gate_kernel(const float* __restrict__ h, const float* __restrict__ g,
                      unsigned short* __restrict__ hn)
{
    int row = blockIdx.x;            // 0 .. 65535
    int tid = threadIdx.x;
    size_t base = (size_t)row * D_DIM;
    float x0 = h[base + tid];
    float x1 = h[base + tid + 256];
    __shared__ float red[256];
    red[tid] = x0 * x0 + x1 * x1;
    __syncthreads();
    for (int s = 128; s > 0; s >>= 1) {
        if (tid < s) red[tid] += red[tid + s];
        __syncthreads();
    }
    float scale = rsqrtf(red[0] * (1.0f / (float)D_DIM) + 1e-6f);
    hn[base + tid]       = f2bf(x0 * scale * g[base + tid]);
    hn[base + tid + 256] = f2bf(x1 * scale * g[base + tid + 256]);
}

// ---------------------------------------------------------------------------
extern "C" void kernel_launch(void* const* d_in, const int* in_sizes, int n_in,
                              void* d_out, int out_size, void* d_ws, size_t ws_size,
                              hipStream_t stream)
{
    const float* x      = (const float*)d_in[0];
    const float* lb     = (const float*)d_in[1];
    const float* W_in   = (const float*)d_in[2];
    const float* b_in   = (const float*)d_in[3];
    const float* W_gate = (const float*)d_in[4];
    const float* b_gate = (const float*)d_in[5];
    const float* W_out  = (const float*)d_in[6];
    const float* b_out  = (const float*)d_in[7];
    float* out = (float*)d_out;

    const size_t ELE = (size_t)M_DIM * D_DIM;            // 33,554,432
    const size_t PSZ = (size_t)B_DIM * 2 * NCH * D_DIM;  // 524,288
    char* ws = (char*)d_ws;
    float*          lam = (float*)(ws);                  // reused as g later
    float*          v   = (float*)(ws + ELE * 4);
    float*          h   = (float*)(ws + ELE * 8);
    unsigned short* hn  = (unsigned short*)(ws + ELE * 12);
    bf16_t*         xb  = (bf16_t*)(ws + ELE * 14);      // bf16 x
    float*          P   = (float*)(ws + ELE * 16);
    float*          S   = P + PSZ;
    float*          C   = S + PSZ;
    bf16_t*         wbi = (bf16_t*)(ws + ELE * 16 + PSZ * 12);            // 512x1024
    bf16_t*         wbg = wbi + (size_t)D_DIM * 2 * D_DIM;                // 512x512
    bf16_t*         wbo = wbg + (size_t)D_DIM * D_DIM;                    // 512x512

    // 0) bf16 pre-conversion (x once; weights tiny, L2-resident afterwards)
    cvt_f32_bf16_kernel<<<(ELE / 4 + 255) / 256, 256, 0, stream>>>(
        x, (unsigned int*)xb, (int)(ELE / 4));
    cvt_f32_bf16_kernel<<<(512 * 1024 / 4 + 255) / 256, 256, 0, stream>>>(
        W_in, (unsigned int*)wbi, 512 * 1024 / 4);
    cvt_f32_bf16_kernel<<<(512 * 512 / 4 + 255) / 256, 256, 0, stream>>>(
        W_gate, (unsigned int*)wbg, 512 * 512 / 4);
    cvt_f32_bf16_kernel<<<(512 * 512 / 4 + 255) / 256, 256, 0, stream>>>(
        W_out, (unsigned int*)wbo, 512 * 512 / 4);

    dim3 gGemm(D_DIM / 64, M_DIM / 128);   // (8, 512)

    // 1) u/f projection -> lam, v   (fused silu / bounded sigmoid epilogue)
    gemm_wmma_kernel<0><<<gGemm, 256, 0, stream>>>(
        xb, wbi, b_in, lb, lam, v, D_DIM, 2 * D_DIM);

    // 2) bidirectional chunked scan
    scan_pass1<<<dim3(1024, 2), 256, 0, stream>>>(lam, v, P, S);
    scan_pass2<<<64, 256, 0, stream>>>(P, S, C);
    scan_pass3<<<1024, 256, 0, stream>>>(lam, v, C, h);

    // 3) gate projection (lam buffer dead after pass3 -> holds g now)
    gemm_wmma_kernel<1><<<gGemm, 256, 0, stream>>>(
        xb, wbg, b_gate, nullptr, lam, nullptr, D_DIM, D_DIM);

    // 4) RMSNorm * gate -> bf16
    norm_gate_kernel<<<M_DIM, 256, 0, stream>>>(h, lam, hn);

    // 5) output projection (bf16 A path)
    gemm_wmma_kernel<2><<<gGemm, 256, 0, stream>>>(
        (const bf16_t*)hn, wbo, b_out, nullptr, out, nullptr, D_DIM, D_DIM);
}
